// GNNPredictor_58763742544938
// MI455X (gfx1250) — compile-verified, matrix operand order checked by reading, not run on hardware
//
#include <hip/hip_runtime.h>
#include <cstdint>

// ---------------------------------------------------------------------------
// Fused EdgeConv x2 for MI455X (gfx1250, wave32, WMMA bf16 16x16x32,
// async global->LDS copies for the gather stage)
// ---------------------------------------------------------------------------

typedef __attribute__((ext_vector_type(16))) __bf16 v16bf;
typedef __attribute__((ext_vector_type(8)))  __bf16 v8bf;
typedef __attribute__((ext_vector_type(8)))  float  v8f;

#define DEVI __device__ __forceinline__

DEVI float bf2f(unsigned short b) {
    return __uint_as_float(((unsigned)b) << 16);
}
DEVI unsigned short f2bf(float f) {
    unsigned u = __float_as_uint(f);
    unsigned lsb = (u >> 16) & 1u;
    return (unsigned short)((u + 0x7FFFu + lsb) >> 16);
}

// sign-aware float atomic max (accumulator pre-initialized to -inf)
DEVI void atomicMaxF(float* p, float v) {
    if (v >= 0.0f) atomicMax((int*)p, __float_as_int(v));
    else           atomicMin((unsigned int*)p, __float_as_uint(v));
}

DEVI v16bf frag16(v8bf lo, v8bf hi) {
    return __builtin_shufflevector(lo, hi, 0,1,2,3,4,5,6,7,8,9,10,11,12,13,14,15);
}

// --------------------------- small prep kernels ----------------------------

__global__ void k_f32_to_bf16(const float* __restrict__ in,
                              unsigned short* __restrict__ out, int n) {
    int i = blockIdx.x * blockDim.x + threadIdx.x;
    if (i < n) out[i] = f2bf(in[i]);
}

__global__ void k_fill_neginf(float* __restrict__ p, int n) {
    int i = blockIdx.x * blockDim.x + threadIdx.x;
    if (i < n) p[i] = __uint_as_float(0xFF800000u);
}

// h1 = bf16(relu(where(finite(agg), agg, 0)))  ==  bf16(max(agg, 0))
__global__ void k_relu_bf16(const float* __restrict__ in,
                            unsigned short* __restrict__ out, int n) {
    int i = blockIdx.x * blockDim.x + threadIdx.x;
    if (i < n) out[i] = f2bf(fmaxf(in[i], 0.0f));
}

// out = where(finite(agg), agg, 0)
__global__ void k_finalize(const float* __restrict__ in,
                           float* __restrict__ out, int n) {
    int i = blockIdx.x * blockDim.x + threadIdx.x;
    if (i < n) {
        float v = in[i];
        out[i] = (v > -__builtin_inff() && v < __builtin_inff()) ? v : 0.0f;
    }
}

// Pack a row-major f32 weight [K=nkb*32][128] into bf16 WMMA B fragments:
// layout out[kb][nt][lane][16], with per-lane K chunks
// lanes 0-15: K {0..7, 16..23}, lanes 16-31: K {8..15, 24..31}, N = nt*16 + (lane&15)
__global__ void k_pack_w(const float* __restrict__ W,
                         unsigned short* __restrict__ out, int nkb) {
    int total = nkb * 4096;                       // nkb*8*32*16
    for (int p = blockIdx.x * blockDim.x + threadIdx.x; p < total;
         p += gridDim.x * blockDim.x) {
        int pos  = p & 15;
        int lane = (p >> 4) & 31;
        int nt   = (p >> 9) & 7;
        int kb   = p >> 12;
        int hi   = lane >> 4;
        int n    = nt * 16 + (lane & 15);
        int kk   = (pos < 8) ? (hi * 8 + pos) : (16 + hi * 8 + (pos - 8));
        out[p] = f2bf(W[(kb * 32 + kk) * 128 + n]);
    }
}

// --------------------------- fused edge kernel -----------------------------
// Workgroup: 128 threads = 4 wave32; 64 edges per workgroup (16 per wave).
// Per wave: M-tile = 16 edges, full N = 128 outputs as 8 WMMA tiles.
// Gather: x_dst half of the message copied with GLOBAL_LOAD_ASYNC_TO_LDS_B128,
//         (x_src - x_dst) half computed with vector b128 loads + ds_store_b128.
// GEMM1 (2C -> 128, +bias, relu) -> LDS -> GEMM2 (128 -> 128, +bias)
//   -> atomic segment-max into agg[dst].

template <int C>
__global__ __launch_bounds__(128) void k_edge_mlp(
    const unsigned short* __restrict__ xb,   // [N][C] bf16 node features
    const int*            __restrict__ ei,   // [2][E] (src row, dst row)
    const unsigned short* __restrict__ waP,  // packed B frags [2C/32][8][32][16]
    const float*          __restrict__ ba,   // [128]
    const unsigned short* __restrict__ wbP,  // packed B frags [4][8][32][16]
    const float*          __restrict__ bb,   // [128]
    float*                __restrict__ agg,  // [N][128], pre-filled -inf
    int E) {
    constexpr int TWO_C = 2 * C;
    constexpr int NKB_A = TWO_C / 32;   // K-steps GEMM1 (4 or 8)
    constexpr int NKB_H = 4;            // K-steps GEMM2 (128/32)
    constexpr int CPE   = C / 8;        // 16B chunks per edge per message half

    __shared__ __align__(16) unsigned short sM[64 * TWO_C]; // edge messages bf16
    __shared__ __align__(16) unsigned short sH[64 * 128];   // hidden bf16
    __shared__ int sDst[64];

    const int tid  = threadIdx.x;
    const int lane = tid & 31;
    const int wave = tid >> 5;
    const int wgE0 = blockIdx.x * 64;

    if (tid < 64) {
        int e = wgE0 + tid;
        sDst[tid] = (e < E) ? ei[E + e] : 0;
    }

    // ---- Phase A: async copy of x_dst rows into the first message half.
    //      sM[e][0..C) <= xb[dst(e)][0..C)   (16B per lane per issue)
    {
        const unsigned long long xbase = (unsigned long long)(uintptr_t)xb;
        #pragma unroll
        for (int ch = tid; ch < 64 * CPE; ch += 128) {
            int e  = ch / CPE;
            int c8 = (ch - e * CPE) * 8;
            int eg = wgE0 + e;
            int d  = (eg < E) ? ei[E + eg] : 0;
            unsigned ldsAddr = (unsigned)(uintptr_t)(sM + e * TWO_C + c8);
            unsigned gOff    = (unsigned)((d * C + c8) * 2);
            asm volatile("global_load_async_to_lds_b128 %0, %1, %2"
                         :: "v"(ldsAddr), "v"(gOff), "s"(xbase)
                         : "memory");
        }
    }

    // ---- Phase B: (x_src - x_dst) into the second message half, vectorized.
    #pragma unroll
    for (int ch = tid; ch < 64 * CPE; ch += 128) {
        int e  = ch / CPE;
        int c8 = (ch - e * CPE) * 8;
        int eg = wgE0 + e;
        int s = 0, d = 0;
        if (eg < E) { s = ei[eg]; d = ei[E + eg]; }
        union U { uint4 v; unsigned short h[8]; } us, ud, ur;
        us.v = *(const uint4*)(xb + s * C + c8);
        ud.v = *(const uint4*)(xb + d * C + c8);
        #pragma unroll
        for (int i = 0; i < 8; ++i)
            ur.h[i] = f2bf(bf2f(us.h[i]) - bf2f(ud.h[i]));
        *(uint4*)(sM + e * TWO_C + C + c8) = ur.v;
    }

    // All of this wave's async LDS writes complete, then workgroup barrier.
    asm volatile("s_wait_asynccnt 0x0" ::: "memory");
    __syncthreads();

    const int m  = lane & 15;
    const int hi = lane >> 4;

    // ---- A fragments for GEMM1 from LDS (row-major, documented 16-bit A layout)
    v16bf aF[NKB_A];
    {
        const unsigned short* row = sM + (wave * 16 + m) * TWO_C;
        #pragma unroll
        for (int kb = 0; kb < NKB_A; ++kb) {
            v8bf c0 = *(const v8bf*)(row + kb * 32 + hi * 8);
            v8bf c1 = *(const v8bf*)(row + kb * 32 + 16 + hi * 8);
            aF[kb] = frag16(c0, c1);
        }
    }

    // ---- GEMM1: [16 x 2C] @ [2C x 128] + ba, relu -> sH (bf16)
    #pragma unroll
    for (int nt = 0; nt < 8; ++nt) {
        float bv = ba[nt * 16 + m];
        v8f acc;
        #pragma unroll
        for (int r = 0; r < 8; ++r) acc[r] = bv;
        #pragma unroll
        for (int kb = 0; kb < NKB_A; ++kb) {
            const unsigned short* bp = waP + ((kb * 8 + nt) * 32 + lane) * 16;
            v8bf b0 = *(const v8bf*)(bp);
            v8bf b1 = *(const v8bf*)(bp + 8);
            v16bf bF = frag16(b0, b1);
            acc = __builtin_amdgcn_wmma_f32_16x16x32_bf16(
                false, aF[kb], false, bF, (short)0, acc, false, false);
        }
        // C layout: VGPR r -> row (hi*8 + r), N = nt*16 + (lane&15)
        #pragma unroll
        for (int r = 0; r < 8; ++r) {
            float v = fmaxf(acc[r], 0.0f);
            sH[(wave * 16 + hi * 8 + r) * 128 + nt * 16 + m] = f2bf(v);
        }
    }

    // ---- A fragments for GEMM2 from sH (wave-local rows; DScnt keeps order)
    v16bf aH[NKB_H];
    {
        const unsigned short* row = sH + (wave * 16 + m) * 128;
        #pragma unroll
        for (int kb = 0; kb < NKB_H; ++kb) {
            v8bf c0 = *(const v8bf*)(row + kb * 32 + hi * 8);
            v8bf c1 = *(const v8bf*)(row + kb * 32 + 16 + hi * 8);
            aH[kb] = frag16(c0, c1);
        }
    }

    int dstRow[8];
    #pragma unroll
    for (int r = 0; r < 8; ++r) dstRow[r] = sDst[wave * 16 + hi * 8 + r];

    // ---- GEMM2: [16 x 128] @ [128 x 128] + bb -> atomic max into agg[dst]
    #pragma unroll
    for (int nt = 0; nt < 8; ++nt) {
        float bv = bb[nt * 16 + m];
        v8f acc;
        #pragma unroll
        for (int r = 0; r < 8; ++r) acc[r] = bv;
        #pragma unroll
        for (int kb = 0; kb < NKB_H; ++kb) {
            const unsigned short* bp = wbP + ((kb * 8 + nt) * 32 + lane) * 16;
            v8bf b0 = *(const v8bf*)(bp);
            v8bf b1 = *(const v8bf*)(bp + 8);
            v16bf bF = frag16(b0, b1);
            acc = __builtin_amdgcn_wmma_f32_16x16x32_bf16(
                false, aH[kb], false, bF, (short)0, acc, false, false);
        }
        int eBase = wgE0 + wave * 16 + hi * 8;
        #pragma unroll
        for (int r = 0; r < 8; ++r) {
            if (eBase + r < E)
                atomicMaxF(agg + dstRow[r] * 128 + nt * 16 + m, acc[r]);
        }
    }
}

// ------------------------------- launcher ----------------------------------

extern "C" void kernel_launch(void* const* d_in, const int* in_sizes, int n_in,
                              void* d_out, int out_size, void* d_ws, size_t ws_size,
                              hipStream_t stream) {
    const float* x   = (const float*)d_in[0];
    const int*   ei  = (const int*)d_in[1];
    const float* W1a = (const float*)d_in[2];
    const float* b1a = (const float*)d_in[3];
    const float* W1b = (const float*)d_in[4];
    const float* b1b = (const float*)d_in[5];
    const float* W2a = (const float*)d_in[6];
    const float* b2a = (const float*)d_in[7];
    const float* W2b = (const float*)d_in[8];
    const float* b2b = (const float*)d_in[9];

    const int IN_CH = 64, HID = 128;
    const int N = in_sizes[0] / IN_CH;   // 50000
    const int E = in_sizes[1] / 2;       // 800000

    // workspace carve-up (256B aligned)
    char* ws = (char*)d_ws;
    size_t off = 0;
    auto take = [&](size_t bytes) {
        void* p = ws + off;
        off += (bytes + 255) & ~(size_t)255;
        return p;
    };
    unsigned short* xb   = (unsigned short*)take((size_t)N * IN_CH * 2);
    unsigned short* h1b  = (unsigned short*)take((size_t)N * HID * 2);
    float*          agg  = (float*)take((size_t)N * HID * 4);
    unsigned short* wa1P = (unsigned short*)take(4  * 4096 * 2);
    unsigned short* wb1P = (unsigned short*)take(4  * 4096 * 2);
    unsigned short* wa2P = (unsigned short*)take(8  * 4096 * 2);
    unsigned short* wb2P = (unsigned short*)take(4  * 4096 * 2);
    (void)ws_size;

    const int T = 256;
    const int nX   = N * IN_CH;
    const int nAgg = N * HID;
    const int nWG  = (E + 63) / 64;

    // prep: x -> bf16, weights -> packed bf16 B fragments
    k_f32_to_bf16<<<(nX + T - 1) / T, T, 0, stream>>>(x, xb, nX);
    k_pack_w<<<(4 * 4096 + T - 1) / T, T, 0, stream>>>(W1a, wa1P, 4);
    k_pack_w<<<(4 * 4096 + T - 1) / T, T, 0, stream>>>(W1b, wb1P, 4);
    k_pack_w<<<(8 * 4096 + T - 1) / T, T, 0, stream>>>(W2a, wa2P, 8);
    k_pack_w<<<(4 * 4096 + T - 1) / T, T, 0, stream>>>(W2b, wb2P, 4);

    // layer 1
    k_fill_neginf<<<(nAgg + T - 1) / T, T, 0, stream>>>(agg, nAgg);
    k_edge_mlp<64><<<nWG, 128, 0, stream>>>(xb, ei, wa1P, b1a, wb1P, b1b, agg, E);
    k_relu_bf16<<<(nAgg + T - 1) / T, T, 0, stream>>>(agg, h1b, nAgg);

    // layer 2
    k_fill_neginf<<<(nAgg + T - 1) / T, T, 0, stream>>>(agg, nAgg);
    k_edge_mlp<128><<<nWG, 128, 0, stream>>>(h1b, ei, wa2P, b2a, wb2P, b2b, agg, E);
    k_finalize<<<(nAgg + T - 1) / T, T, 0, stream>>>(agg, (float*)d_out, nAgg);
}
